// GRU_68470368633225
// MI455X (gfx1250) — matrix operand order repaired and checked by reading, not compiled, not run on hardware
//
#include <hip/hip_runtime.h>
#include <cstdint>
#include <cstddef>

#define Tt 64
#define Bb 64
#define Ee 512
#define Hh 1024
#define Vv 10000
#define Ll 2

typedef __attribute__((ext_vector_type(16))) __bf16 bf16x16;
typedef __attribute__((ext_vector_type(8)))  __bf16 bf16x8;
typedef __attribute__((ext_vector_type(8)))  float  f32x8;
typedef __attribute__((ext_vector_type(4)))  unsigned int u32x4;
typedef __attribute__((ext_vector_type(8)))  int i32x8;
typedef __attribute__((ext_vector_type(4)))  int i32x4;

__device__ __forceinline__ __bf16 f2bf(float f) {
  union { float f; uint32_t u; } x; x.f = f;
  uint32_t r = x.u + 0x7FFFu + ((x.u >> 16) & 1u);   // round-to-nearest-even
  union { uint16_t u; __bf16 b; } y; y.u = (uint16_t)(r >> 16);
  return y.b;
}

__device__ __forceinline__ float sigm(float x) { return 1.0f / (1.0f + __expf(-x)); }

// ---------------------------------------------------------- TDM panel stage
// 2-D Tensor DMA descriptor (D#): data_size=2B, tile = 16 rows x cols elems,
// contiguous (stride == cols). Groups 2/3 NULL-equivalent (zero).
__device__ __forceinline__ void tdm_load_2d(uint32_t lds_addr, const void* gaddr,
                                            uint32_t cols) {
  uint64_t ga = (uint64_t)(uintptr_t)gaddr;
  u32x4 g0;
  g0[0] = 1u;                                          // count=1 (valid), user mode
  g0[1] = lds_addr;                                    // LDS byte address
  g0[2] = (uint32_t)(ga & 0xFFFFFFFFu);                // global_addr[31:0]
  g0[3] = (uint32_t)((ga >> 32) & 0x1FFFFFFu) | (2u << 30);  // addr[56:32], type=2
  i32x8 g1;
  g1[0] = (int)(1u << 16);                             // data_size = 2 bytes
  g1[1] = (int)((cols & 0xFFFFu) << 16);               // tensor_dim0[15:0]
  g1[2] = (int)(((cols >> 16) & 0xFFFFu) | (16u << 16));     // dim0 hi | tensor_dim1=16
  g1[3] = (int)((cols & 0xFFFFu) << 16);               // tensor_dim1 hi=0 | tile_dim0=cols
  g1[4] = (int)16u;                                    // tile_dim1=16, tile_dim2=0
  g1[5] = (int)cols;                                   // tensor_dim0_stride[31:0]
  g1[6] = 0;                                           // stride hi, dim1_stride lo
  g1[7] = 0;
  i32x4 z4 = {0, 0, 0, 0};
#if __has_builtin(__builtin_amdgcn_tensor_load_to_lds)
#if __clang_major__ >= 23
  i32x8 z8 = {0, 0, 0, 0, 0, 0, 0, 0};
  __builtin_amdgcn_tensor_load_to_lds(g0, g1, z4, z4, z8, 0);
#else
  __builtin_amdgcn_tensor_load_to_lds(g0, g1, z4, z4, 0);
#endif
#endif
}

// Stage a contiguous 16-row x cols bf16 panel into LDS. Caller must __syncthreads().
__device__ __forceinline__ void stage16(__bf16* sdst, const __bf16* gsrc, int cols) {
#if __has_builtin(__builtin_amdgcn_tensor_load_to_lds)
  if ((threadIdx.x >> 5) == 0) {
    tdm_load_2d((uint32_t)(uintptr_t)sdst, gsrc, (uint32_t)cols);
    __builtin_amdgcn_s_wait_tensorcnt(0);
  }
#else
  const uint4* s = (const uint4*)gsrc;
  uint4* d = (uint4*)sdst;
  int n = (16 * cols) >> 3;                     // 8 bf16 per uint4
  for (int i = threadIdx.x; i < n; i += (int)blockDim.x) d[i] = s[i];
#endif
}

// ----------------------------------------------------------- WMMA fragments
// A fragment from an LDS panel [16, LD] (row = lane%16).
// Lanes 0-15 hold K={0..7,16..23}, lanes 16-31 hold K={8..15,24..31}.
template <int LD>
__device__ __forceinline__ bf16x16 load_a_sh(const __bf16* sA, int kb) {
  int lane = threadIdx.x & 31;
  int kg = lane >> 4;
  const __bf16* p = sA + (lane & 15) * LD + kb + 8 * kg;
  bf16x8 lo = *(const bf16x8*)p;
  bf16x8 hi = *(const bf16x8*)(p + 16);
  bf16x16 a;
#pragma unroll
  for (int i = 0; i < 8; ++i) { a[i] = lo[i]; a[i + 8] = hi[i]; }
  return a;
}

// B fragment: 32x16 (KxN) tile of W^T where W is row-major [N, ldw].
// Column n of B == row n of W: lanes 0-15 hold K={0..7,16..23}, lanes 16-31 hold K={8..15,24..31}.
__device__ __forceinline__ bf16x16 load_b(const __bf16* W, int ldw, int nt, int kb) {
  int lane = threadIdx.x & 31;
  int kg = lane >> 4;
  const __bf16* p = W + (size_t)(nt * 16 + (lane & 15)) * ldw + kb + 8 * kg;
  bf16x8 lo = *(const bf16x8*)p;
  bf16x8 hi = *(const bf16x8*)(p + 16);
  bf16x16 b;
#pragma unroll
  for (int i = 0; i < 8; ++i) { b[i] = lo[i]; b[i + 8] = hi[i]; }
  return b;
}

__device__ __forceinline__ f32x8 wmma_bf16(bf16x16 a, bf16x16 b, f32x8 c) {
  return __builtin_amdgcn_wmma_f32_16x16x32_bf16(false, a, false, b, (short)0, c, false, false);
}

// ---------------------------------------------------------------- utilities
__global__ void k_cvt(const float* __restrict__ src, __bf16* __restrict__ dst, int n) {
  int i = blockIdx.x * blockDim.x + threadIdx.x;
  if (i < n) dst[i] = f2bf(src[i]);
}

__global__ void k_embed(const int* __restrict__ inputs, const float* __restrict__ emb,
                        __bf16* __restrict__ X) {
  int row = blockIdx.x;                    // 0 .. T*B-1
  int v = inputs[row];
  for (int c = threadIdx.x; c < Ee; c += blockDim.x)
    X[(size_t)row * Ee + c] = f2bf(emb[(size_t)v * Ee + c]);
}

__global__ void k_init_state(const float* __restrict__ hidden,
                             float* __restrict__ H0F, __bf16* __restrict__ H0B,
                             float* __restrict__ H1F, __bf16* __restrict__ H1B) {
  int i = blockIdx.x * blockDim.x + threadIdx.x;
  if (i < Bb * Hh) {
    float a = hidden[i];
    float b = hidden[Bb * Hh + i];
    H0F[i] = a; H0B[i] = f2bf(a);
    H1F[i] = b; H1B[i] = f2bf(b);
  }
}

__global__ void k_final_state(const float* __restrict__ H0F, const float* __restrict__ H1F,
                              float* __restrict__ out) {
  int i = blockIdx.x * blockDim.x + threadIdx.x;
  if (i < Bb * Hh) { out[i] = H0F[i]; out[Bb * Hh + i] = H1F[i]; }
}

// ------------------------------------------------ layer-0 input projections
// OUT[T*B, H] = X[T*B, E] @ W[H, E]^T   (mt uniform per block: 8 waves/block)
__global__ void k_proj0(const __bf16* __restrict__ X, const __bf16* __restrict__ W,
                        float* __restrict__ OUT) {
  __shared__ __align__(16) __bf16 sA[16 * Ee];
  int warp = threadIdx.x >> 5;
  int mt = blockIdx.x >> 3;                       // 0..255 (uniform in block)
  int nt = (blockIdx.x * 8 + warp) & 63;          // 0..63
  stage16(sA, X + (size_t)mt * 16 * Ee, Ee);
  __syncthreads();
  f32x8 acc = {};
  for (int kb = 0; kb < Ee; kb += 32) {
    bf16x16 a = load_a_sh<Ee>(sA, kb);
    bf16x16 b = load_b(W, Ee, nt, kb);
    acc = wmma_bf16(a, b, acc);
  }
  int lane = threadIdx.x & 31, kg = lane >> 4, n = lane & 15;
#pragma unroll
  for (int j = 0; j < 8; ++j) {
    int m = mt * 16 + j + 8 * kg;
    OUT[(size_t)m * Hh + nt * 16 + n] = acc[j];
  }
}

// ------------------------------------------------------- layer-0 GRU cell
__global__ void k_l0_rz(const float* __restrict__ RXt, const float* __restrict__ ZXt,
                        const float* __restrict__ H0F, const __bf16* __restrict__ H0B,
                        const __bf16* __restrict__ Ur0, const __bf16* __restrict__ Uz0,
                        const float* __restrict__ bur0, const float* __restrict__ buz0,
                        float* __restrict__ Zbuf, __bf16* __restrict__ RH) {
  __shared__ __align__(16) __bf16 sA[16 * Hh];
  int warp = threadIdx.x >> 5;
  int mt = blockIdx.x >> 3;                       // 0..3 (uniform in block)
  int nt = (blockIdx.x * 8 + warp) & 63;          // 0..63
  stage16(sA, H0B + (size_t)mt * 16 * Hh, Hh);
  __syncthreads();
  int lane = threadIdx.x & 31, kg = lane >> 4, n = lane & 15;
  int ng = nt * 16 + n;
  float br = bur0[ng], bz = buz0[ng];
  f32x8 ar, az;
#pragma unroll
  for (int j = 0; j < 8; ++j) {
    int m = mt * 16 + j + 8 * kg;
    ar[j] = RXt[(size_t)m * Hh + ng] + br;
    az[j] = ZXt[(size_t)m * Hh + ng] + bz;
  }
  for (int kb = 0; kb < Hh; kb += 32) {
    bf16x16 a   = load_a_sh<Hh>(sA, kb);
    bf16x16 br_ = load_b(Ur0, Hh, nt, kb);
    ar = wmma_bf16(a, br_, ar);
    bf16x16 bz_ = load_b(Uz0, Hh, nt, kb);
    az = wmma_bf16(a, bz_, az);
  }
#pragma unroll
  for (int j = 0; j < 8; ++j) {
    int m = mt * 16 + j + 8 * kg;
    float r = sigm(ar[j]);
    float z = sigm(az[j]);
    float h0 = H0F[(size_t)m * Hh + ng];
    Zbuf[(size_t)m * Hh + ng] = z;
    RH[(size_t)m * Hh + ng] = f2bf(r * h0);
  }
}

__global__ void k_l0_h(const float* __restrict__ HXt, const __bf16* __restrict__ RH,
                       const __bf16* __restrict__ Uh0, const float* __restrict__ buh0,
                       const float* __restrict__ Zbuf,
                       float* __restrict__ H0F, __bf16* __restrict__ H0B) {
  __shared__ __align__(16) __bf16 sA[16 * Hh];
  int warp = threadIdx.x >> 5;
  int mt = blockIdx.x >> 3;
  int nt = (blockIdx.x * 8 + warp) & 63;
  stage16(sA, RH + (size_t)mt * 16 * Hh, Hh);
  __syncthreads();
  int lane = threadIdx.x & 31, kg = lane >> 4, n = lane & 15;
  int ng = nt * 16 + n;
  float bh = buh0[ng];
  f32x8 acc;
#pragma unroll
  for (int j = 0; j < 8; ++j) {
    int m = mt * 16 + j + 8 * kg;
    acc[j] = HXt[(size_t)m * Hh + ng] + bh;
  }
  for (int kb = 0; kb < Hh; kb += 32) {
    bf16x16 a = load_a_sh<Hh>(sA, kb);
    bf16x16 b = load_b(Uh0, Hh, nt, kb);
    acc = wmma_bf16(a, b, acc);
  }
#pragma unroll
  for (int j = 0; j < 8; ++j) {
    int m = mt * 16 + j + 8 * kg;
    size_t idx = (size_t)m * Hh + ng;
    float ht = tanhf(acc[j]);
    float z = Zbuf[idx];
    float hn = (1.0f - z) * H0F[idx] + z * ht;
    H0F[idx] = hn;
    H0B[idx] = f2bf(hn);
  }
}

// ------------------------------------------------------- layer-1 GRU cell
__global__ void k_l1_rz(const __bf16* __restrict__ X /* = H0B */,
                        const float* __restrict__ H1F, const __bf16* __restrict__ H1B,
                        const __bf16* __restrict__ W1r, const __bf16* __restrict__ W1z,
                        const __bf16* __restrict__ W1h,
                        const __bf16* __restrict__ Ur1, const __bf16* __restrict__ Uz1,
                        const float* __restrict__ brR, const float* __restrict__ bzR,
                        const float* __restrict__ bhR,
                        const float* __restrict__ bur1, const float* __restrict__ buz1,
                        float* __restrict__ Zbuf, __bf16* __restrict__ RH,
                        float* __restrict__ HX1) {
  __shared__ __align__(16) __bf16 sX[16 * Hh];
  __shared__ __align__(16) __bf16 sH[16 * Hh];
  int warp = threadIdx.x >> 5;
  int mt = blockIdx.x >> 3;
  int nt = (blockIdx.x * 8 + warp) & 63;
  stage16(sX, X   + (size_t)mt * 16 * Hh, Hh);
  stage16(sH, H1B + (size_t)mt * 16 * Hh, Hh);
  __syncthreads();
  int lane = threadIdx.x & 31, kg = lane >> 4, n = lane & 15;
  int ng = nt * 16 + n;
  float br = brR[ng] + bur1[ng];
  float bz = bzR[ng] + buz1[ng];
  float bh = bhR[ng];
  f32x8 ar, az, ah;
#pragma unroll
  for (int j = 0; j < 8; ++j) { ar[j] = br; az[j] = bz; ah[j] = bh; }
  for (int kb = 0; kb < Hh; kb += 32) {
    bf16x16 ax = load_a_sh<Hh>(sX, kb);
    bf16x16 b0 = load_b(W1r, Hh, nt, kb);
    ar = wmma_bf16(ax, b0, ar);
    bf16x16 b1 = load_b(W1z, Hh, nt, kb);
    az = wmma_bf16(ax, b1, az);
    bf16x16 b2 = load_b(W1h, Hh, nt, kb);
    ah = wmma_bf16(ax, b2, ah);
    bf16x16 ah1 = load_a_sh<Hh>(sH, kb);
    bf16x16 b3 = load_b(Ur1, Hh, nt, kb);
    ar = wmma_bf16(ah1, b3, ar);
    bf16x16 b4 = load_b(Uz1, Hh, nt, kb);
    az = wmma_bf16(ah1, b4, az);
  }
#pragma unroll
  for (int j = 0; j < 8; ++j) {
    int m = mt * 16 + j + 8 * kg;
    size_t idx = (size_t)m * Hh + ng;
    float r = sigm(ar[j]);
    float z = sigm(az[j]);
    float h1 = H1F[idx];
    Zbuf[idx] = z;
    RH[idx] = f2bf(r * h1);
    HX1[idx] = ah[j];
  }
}

__global__ void k_l1_h(int t, const float* __restrict__ HX1, const __bf16* __restrict__ RH,
                       const __bf16* __restrict__ Uh1, const float* __restrict__ buh1,
                       const float* __restrict__ Zbuf,
                       float* __restrict__ H1F, __bf16* __restrict__ H1B,
                       __bf16* __restrict__ H1SEQ) {
  __shared__ __align__(16) __bf16 sA[16 * Hh];
  int warp = threadIdx.x >> 5;
  int mt = blockIdx.x >> 3;
  int nt = (blockIdx.x * 8 + warp) & 63;
  stage16(sA, RH + (size_t)mt * 16 * Hh, Hh);
  __syncthreads();
  int lane = threadIdx.x & 31, kg = lane >> 4, n = lane & 15;
  int ng = nt * 16 + n;
  float bh = buh1[ng];
  f32x8 acc;
#pragma unroll
  for (int j = 0; j < 8; ++j) acc[j] = HX1[(size_t)(mt * 16 + j + 8 * kg) * Hh + ng] + bh;
  for (int kb = 0; kb < Hh; kb += 32) {
    bf16x16 a = load_a_sh<Hh>(sA, kb);
    bf16x16 b = load_b(Uh1, Hh, nt, kb);
    acc = wmma_bf16(a, b, acc);
  }
#pragma unroll
  for (int j = 0; j < 8; ++j) {
    int m = mt * 16 + j + 8 * kg;
    size_t idx = (size_t)m * Hh + ng;
    float ht = tanhf(acc[j]);
    float z = Zbuf[idx];
    float hn = (1.0f - z) * H1F[idx] + z * ht;
    H1F[idx] = hn;
    __bf16 hb = f2bf(hn);
    H1B[idx] = hb;
    H1SEQ[((size_t)t * Bb + m) * Hh + ng] = hb;
  }
}

// ----------------------------------------------------------- output logits
// OUT[4096, 10000] = H1SEQ[4096, 1024] @ Wout[10000, 1024]^T + bout
// grid: (79, 256); blockIdx.y = mt, nt = blockIdx.x*8 + warp (clamped, 625 tiles).
__global__ void k_logits(const __bf16* __restrict__ H1SEQ, const __bf16* __restrict__ Woutb,
                         const float* __restrict__ bout, float* __restrict__ OUT) {
  __shared__ __align__(16) __bf16 sA[16 * Hh];
  int warp = threadIdx.x >> 5;
  int mt = blockIdx.y;                            // 0..255 (uniform in block)
  int nt_raw = blockIdx.x * 8 + warp;             // 0..631
  int nt = nt_raw < 625 ? nt_raw : 624;           // clamp: keep EXEC all-ones
  stage16(sA, H1SEQ + (size_t)mt * 16 * Hh, Hh);
  __syncthreads();
  int lane = threadIdx.x & 31, kg = lane >> 4, n = lane & 15;
  int ng = nt * 16 + n;
  float b0 = bout[ng];
  f32x8 acc;
#pragma unroll
  for (int j = 0; j < 8; ++j) acc[j] = b0;
  const __bf16* wrow = Woutb + (size_t)(nt * 16 + (lane & 15)) * Hh;
  for (int kb = 0; kb < Hh; kb += 32) {
    __builtin_prefetch(wrow + kb + 64, 0, 1);     // stream Wout rows ahead
    bf16x16 a = load_a_sh<Hh>(sA, kb);
    bf16x16 b = load_b(Woutb, Hh, nt, kb);
    acc = wmma_bf16(a, b, acc);
  }
  if (nt_raw < 625) {
#pragma unroll
    for (int j = 0; j < 8; ++j) {
      int m = mt * 16 + j + 8 * kg;
      OUT[(size_t)m * Vv + ng] = acc[j];
    }
  }
}

// ---------------------------------------------------------------- launcher
extern "C" void kernel_launch(void* const* d_in, const int* in_sizes, int n_in,
                              void* d_out, int out_size, void* d_ws, size_t ws_size,
                              hipStream_t stream) {
  (void)in_sizes; (void)n_in; (void)out_size; (void)ws_size;

  const int*   inputs = (const int*)d_in[0];
  const float* hidden = (const float*)d_in[1];
  const float* emb    = (const float*)d_in[2];
  const float* brR  = (const float*)d_in[9];
  const float* bzR  = (const float*)d_in[10];
  const float* bhR  = (const float*)d_in[11];
  const float* bur  = (const float*)d_in[15];
  const float* buz  = (const float*)d_in[16];
  const float* buh  = (const float*)d_in[17];
  const float* bout = (const float*)d_in[19];

  char* wsb = (char*)d_ws;
  size_t off = 0;
  auto alloc = [&](size_t bytes) -> void* {
    void* p = wsb + off;
    off = (off + bytes + 255) & ~(size_t)255;
    return p;
  };

  __bf16* XEMB  = (__bf16*)alloc((size_t)Tt * Bb * Ee * 2);
  __bf16* W0R   = (__bf16*)alloc((size_t)Hh * Ee * 2);
  __bf16* W0Z   = (__bf16*)alloc((size_t)Hh * Ee * 2);
  __bf16* W0H   = (__bf16*)alloc((size_t)Hh * Ee * 2);
  __bf16* W1R   = (__bf16*)alloc((size_t)Hh * Hh * 2);
  __bf16* W1Z   = (__bf16*)alloc((size_t)Hh * Hh * 2);
  __bf16* W1H   = (__bf16*)alloc((size_t)Hh * Hh * 2);
  __bf16* URB   = (__bf16*)alloc((size_t)Ll * Hh * Hh * 2);
  __bf16* UZB   = (__bf16*)alloc((size_t)Ll * Hh * Hh * 2);
  __bf16* UHB   = (__bf16*)alloc((size_t)Ll * Hh * Hh * 2);
  __bf16* WOUTB = (__bf16*)alloc((size_t)Vv * Hh * 2);
  float*  RX0   = (float*)alloc((size_t)Tt * Bb * Hh * 4);
  float*  ZX0   = (float*)alloc((size_t)Tt * Bb * Hh * 4);
  float*  HX0   = (float*)alloc((size_t)Tt * Bb * Hh * 4);
  float*  H0F   = (float*)alloc((size_t)Bb * Hh * 4);
  __bf16* H0B   = (__bf16*)alloc((size_t)Bb * Hh * 2);
  float*  H1F   = (float*)alloc((size_t)Bb * Hh * 4);
  __bf16* H1B   = (__bf16*)alloc((size_t)Bb * Hh * 2);
  float*  Z0    = (float*)alloc((size_t)Bb * Hh * 4);
  __bf16* RH0   = (__bf16*)alloc((size_t)Bb * Hh * 2);
  float*  Z1    = (float*)alloc((size_t)Bb * Hh * 4);
  __bf16* RH1   = (__bf16*)alloc((size_t)Bb * Hh * 2);
  float*  HX1   = (float*)alloc((size_t)Bb * Hh * 4);
  __bf16* H1SEQ = (__bf16*)alloc((size_t)Tt * Bb * Hh * 2);

  auto cvt = [&](int in_idx, __bf16* dst, size_t n) {
    k_cvt<<<(unsigned)((n + 255) / 256), 256, 0, stream>>>((const float*)d_in[in_idx], dst, (int)n);
  };
  cvt(3,  W0R,   (size_t)Hh * Ee);
  cvt(4,  W0Z,   (size_t)Hh * Ee);
  cvt(5,  W0H,   (size_t)Hh * Ee);
  cvt(6,  W1R,   (size_t)Hh * Hh);
  cvt(7,  W1Z,   (size_t)Hh * Hh);
  cvt(8,  W1H,   (size_t)Hh * Hh);
  cvt(12, URB,   (size_t)Ll * Hh * Hh);
  cvt(13, UZB,   (size_t)Ll * Hh * Hh);
  cvt(14, UHB,   (size_t)Ll * Hh * Hh);
  cvt(18, WOUTB, (size_t)Vv * Hh);

  k_embed<<<Tt * Bb, 256, 0, stream>>>(inputs, emb, XEMB);
  k_init_state<<<(Bb * Hh + 255) / 256, 256, 0, stream>>>(hidden, H0F, H0B, H1F, H1B);

  // layer-0 input projections for all timesteps at once
  k_proj0<<<2048, 256, 0, stream>>>(XEMB, W0R, RX0);
  k_proj0<<<2048, 256, 0, stream>>>(XEMB, W0Z, ZX0);
  k_proj0<<<2048, 256, 0, stream>>>(XEMB, W0H, HX0);

  const __bf16* Ur0 = URB;              const __bf16* Ur1 = URB + (size_t)Hh * Hh;
  const __bf16* Uz0 = UZB;              const __bf16* Uz1 = UZB + (size_t)Hh * Hh;
  const __bf16* Uh0 = UHB;              const __bf16* Uh1 = UHB + (size_t)Hh * Hh;
  const float* bur0 = bur;              const float* bur1 = bur + Hh;
  const float* buz0 = buz;              const float* buz1 = buz + Hh;
  const float* buh0 = buh;              const float* buh1 = buh + Hh;

  for (int t = 0; t < Tt; ++t) {
    const float* RXt = RX0 + (size_t)t * Bb * Hh;
    const float* ZXt = ZX0 + (size_t)t * Bb * Hh;
    const float* HXt = HX0 + (size_t)t * Bb * Hh;
    k_l0_rz<<<32, 256, 0, stream>>>(RXt, ZXt, H0F, H0B, Ur0, Uz0, bur0, buz0, Z0, RH0);
    k_l0_h<<<32, 256, 0, stream>>>(HXt, RH0, Uh0, buh0, Z0, H0F, H0B);
    k_l1_rz<<<32, 256, 0, stream>>>(H0B, H1F, H1B, W1R, W1Z, W1H, Ur1, Uz1,
                                    brR, bzR, bhR, bur1, buz1, Z1, RH1, HX1);
    k_l1_h<<<32, 256, 0, stream>>>(t, HX1, RH1, Uh1, buh1, Z1, H1F, H1B, H1SEQ);
  }

  float* logits = (float*)d_out;
  dim3 lgrid(79, 256);
  k_logits<<<lgrid, 256, 0, stream>>>(H1SEQ, WOUTB, bout, logits);

  float* hfinal = logits + (size_t)Tt * Bb * Vv;
  k_final_state<<<(Bb * Hh + 255) / 256, 256, 0, stream>>>(H0F, H1F, hfinal);
}
